// FlashAttention_84112639525332
// MI455X (gfx1250) — matrix operand order repaired and checked by reading, not compiled, not run on hardware
//
#include <hip/hip_runtime.h>

typedef __attribute__((ext_vector_type(16))) _Float16 v16h;
typedef __attribute__((ext_vector_type(8)))  _Float16 v8h;
typedef __attribute__((ext_vector_type(2)))  _Float16 v2h;
typedef __attribute__((ext_vector_type(8)))  float    v8f;
typedef __attribute__((ext_vector_type(4)))  float    f4;

constexpr int Bb = 2;
constexpr int Ll = 2048;
constexpr int Hh = 16;
constexpr int Dd = 64;
constexpr int NK = 32;      // keys per block-iteration
constexpr int KSTRH = 72;   // LDS K row stride (halves): 144B, 16B-aligned
constexpr int VSTRH = 40;   // LDS V^T row stride (halves): 80B, 16B-aligned

// log2(e); softmax computed in the log2 domain so v_exp_f32 needs no pre-mul
constexpr float LOG2E   = 1.44269504088896340736f;
constexpr float QSCALE  = 0.125f * LOG2E;          // 1/sqrt(64) * log2(e)
constexpr float NEGBIAS = -1.4426950408e10f;       // -1e10 * log2(e)

#define WMMA16(a, b, c) \
  __builtin_amdgcn_wmma_f32_16x16x32_f16(false, (a), false, (b), (short)0, (c), false, false)

__device__ inline float fast_exp2(float x) {
#if __has_builtin(__builtin_amdgcn_exp2f)
  return __builtin_amdgcn_exp2f(x);
#else
  return __expf(x * 0.6931471805599453f);
#endif
}

// Swap a dword with the opposite 16-lane half (xor-16) in one VALU op.
__device__ inline int xor16i(int x) {
  return __builtin_amdgcn_permlanex16(x, x, 0x76543210, 0xfedcba98, false, false);
}
__device__ inline float xor16f(float x) {
  return __int_as_float(xor16i(__float_as_int(x)));
}

// block permutation for 8-element groups: [0,2,1,3] pattern (swap middle pair)
__device__ inline int blkperm(int blk) {
  int b3 = blk & 3;
  return (b3 == 1 || b3 == 2) ? (blk ^ 3) : blk;
}

// WMMA 16-bit operand = 16 contiguous halves in LDS (k = 16h + s).
__device__ inline v16h ld_frag16(const _Float16* p) {
  union { v16h v; v8h h[2]; } u;
  u.h[0] = *(const v8h*)(p);
  u.h[1] = *(const v8h*)(p + 8);
  return u.v;
}

// B-operand fragment from global f32 (16 contiguous elements), with scale
__device__ inline v16h cvt_b_frag(const float* p, float sc) {
  f4 r0 = *(const f4*)(p);
  f4 r1 = *(const f4*)(p + 4);
  f4 r2 = *(const f4*)(p + 8);
  f4 r3 = *(const f4*)(p + 12);
  v16h b;
  b[0]  = (_Float16)(r0.x * sc); b[1]  = (_Float16)(r0.y * sc);
  b[2]  = (_Float16)(r0.z * sc); b[3]  = (_Float16)(r0.w * sc);
  b[4]  = (_Float16)(r1.x * sc); b[5]  = (_Float16)(r1.y * sc);
  b[6]  = (_Float16)(r1.z * sc); b[7]  = (_Float16)(r1.w * sc);
  b[8]  = (_Float16)(r2.x * sc); b[9]  = (_Float16)(r2.y * sc);
  b[10] = (_Float16)(r2.z * sc); b[11] = (_Float16)(r2.w * sc);
  b[12] = (_Float16)(r3.x * sc); b[13] = (_Float16)(r3.y * sc);
  b[14] = (_Float16)(r3.z * sc); b[15] = (_Float16)(r3.w * sc);
  return b;
}

// Stage one 32-key K/V block into LDS (f16, natural d/key order, V transposed).
__device__ inline void stage_block(const float* __restrict__ K,
                                   const float* __restrict__ V,
                                   const int*   __restrict__ Msk,
                                   _Float16* __restrict__ dK,
                                   _Float16* __restrict__ dVt,
                                   float*    __restrict__ dMB,
                                   int tid, size_t gbase, int mbase) {
  {  // K: one thread = one key row, one 8-d block -> single b128 store
    int key = tid >> 3;
    int blk = tid & 7;
    const float* g = K + gbase + (size_t)key * (Hh * Dd) + blk * 8;
    f4 k0 = *(const f4*)(g);
    f4 k1 = *(const f4*)(g + 4);
    v8h kh;
    kh[0] = (_Float16)k0.x; kh[1] = (_Float16)k0.y;
    kh[2] = (_Float16)k0.z; kh[3] = (_Float16)k0.w;
    kh[4] = (_Float16)k1.x; kh[5] = (_Float16)k1.y;
    kh[6] = (_Float16)k1.z; kh[7] = (_Float16)k1.w;
    *(v8h*)&dK[key * KSTRH + blk * 8] = kh;
  }
  {  // V^T: one thread = two adjacent keys x 4 d's -> 4 packed-pair b32 stores
    int key0 = (tid >> 4) * 2;
    int d4   = (tid & 15) << 2;
    const float* g0 = V + gbase + (size_t)key0 * (Hh * Dd) + d4;
    f4 v0 = *(const f4*)(g0);
    f4 v1 = *(const f4*)(g0 + Hh * Dd);
    int pkey = key0;  // even
    v2h p0; p0[0] = (_Float16)v0.x; p0[1] = (_Float16)v1.x;
    v2h p1; p1[0] = (_Float16)v0.y; p1[1] = (_Float16)v1.y;
    v2h p2; p2[0] = (_Float16)v0.z; p2[1] = (_Float16)v1.z;
    v2h p3; p3[0] = (_Float16)v0.w; p3[1] = (_Float16)v1.w;
    *(v2h*)&dVt[(d4 + 0) * VSTRH + pkey] = p0;
    *(v2h*)&dVt[(d4 + 1) * VSTRH + pkey] = p1;
    *(v2h*)&dVt[(d4 + 2) * VSTRH + pkey] = p2;
    *(v2h*)&dVt[(d4 + 3) * VSTRH + pkey] = p3;
  }
  if (tid < NK)
    dMB[tid] = (Msk[mbase + tid] > 0) ? 0.0f : NEGBIAS;
}

__global__ __launch_bounds__(256) void fa_fwd_wmma(const float* __restrict__ Q,
                                                   const float* __restrict__ K,
                                                   const float* __restrict__ V,
                                                   const int*   __restrict__ Msk,
                                                   float* __restrict__ O) {
  __shared__ __align__(16) _Float16 ldsK[2][NK * KSTRH];   // f16 K block [key][d]
  __shared__ __align__(16) _Float16 ldsVt[2][Dd * VSTRH];  // f16 V^T block [d][key]
  __shared__ __align__(16) float    ldsMB[2][NK];          // additive mask bias (log2 dom)

  const int tid  = threadIdx.x;
  const int lane = tid & 31;
  const int wave = tid >> 5;
  const int col  = lane & 15;   // tile row/col owned by this lane
  const int half = lane >> 4;   // lane half -> K-dim element group

  const int qblocks = Ll / 128;
  const int b    = blockIdx.x / (Hh * qblocks);
  const int rem  = blockIdx.x % (Hh * qblocks);
  const int head = rem / qblocks;
  const int q0   = (rem % qblocks) * 128 + wave * 16;

  // ---- Q^T B-operand fragments, loaded once (scale folds 1/sqrt(D)*log2e) ----
  const float* qrow = Q + ((size_t)(b * Ll + q0 + col) * Hh + head) * Dd;
  v16h bQ[2];
#pragma unroll
  for (int s = 0; s < 2; ++s)
    bQ[s] = cvt_b_frag(qrow + s * 32 + 16 * half, QSCALE);

  v8f acc[4];
#pragma unroll
  for (int c = 0; c < 4; ++c) acc[c] = v8f{};
  float mrun = -1e30f;
  float lrun = 0.0f;

  const size_t gkv0 = ((size_t)(b * Ll) * Hh + head) * Dd;

  stage_block(K, V, Msk, ldsK[0], ldsVt[0], ldsMB[0], tid, gkv0, b * Ll);

  for (int kb = 0; kb < Ll; kb += NK) {
    const int cur = (kb / NK) & 1;
    __syncthreads();

    // issue next block's staging before compute (double buffer hides latency)
    if (kb + NK < Ll)
      stage_block(K, V, Msk, ldsK[cur ^ 1], ldsVt[cur ^ 1], ldsMB[cur ^ 1], tid,
                  gkv0 + (size_t)(kb + NK) * (Hh * Dd), b * Ll + kb + NK);
    if (kb + 2 * NK < Ll) {  // prefetch two blocks ahead into cache
      int key = tid >> 4;
      int d4  = (tid & 15) << 2;
      size_t goff = gkv0 + (size_t)(kb + 2 * NK + key) * (Hh * Dd) + d4;
      __builtin_prefetch(K + goff, 0, 1);
      __builtin_prefetch(V + goff, 0, 1);
    }

    // ---- S^T tiles: st[t] = K_tile(t).Q^T + mask_bias (bias seeds C) ----
    v8f st[2];
#pragma unroll
    for (int t = 0; t < 2; ++t) {
      const float* mb = &ldsMB[cur][t * 16 + half * 8];
      f4 m0 = *(const f4*)mb;
      f4 m1 = *(const f4*)(mb + 4);
      v8f c;
      c[0] = m0.x; c[1] = m0.y; c[2] = m0.z; c[3] = m0.w;
      c[4] = m1.x; c[5] = m1.y; c[6] = m1.z; c[7] = m1.w;
      const _Float16* kr = &ldsK[cur][(t * 16 + col) * KSTRH];
#pragma unroll
      for (int s = 0; s < 2; ++s) {
        v16h aK = ld_frag16(kr + s * 32 + half * 16);  // contiguous operand
        c = WMMA16(aK, bQ[s], c);
      }
      st[t] = c;
    }

    // ---- online softmax, log2 domain (per-lane scalar stats: lane == query) ----
    float tmax = -1e30f;
#pragma unroll
    for (int i = 0; i < 8; ++i) {
      tmax = fmaxf(tmax, st[0][i]);
      tmax = fmaxf(tmax, st[1][i]);
    }
    tmax = fmaxf(tmax, xor16f(tmax));
    float mnew  = fmaxf(mrun, tmax);
    float alpha = fast_exp2(mrun - mnew);

    float p0[8], p1[8];
    float sum = 0.0f;
#pragma unroll
    for (int i = 0; i < 8; ++i) {
      p0[i] = fast_exp2(st[0][i] - mnew);
      p1[i] = fast_exp2(st[1][i] - mnew);
      sum += p0[i] + p1[i];
    }
    sum += xor16f(sum);
    lrun = lrun * alpha + sum;
    mrun = mnew;

    // ---- pack P^T to f16 dwords, then exchange/select packed (xor-16) ----
    union { v16h v; int i[8]; } up;
#pragma unroll
    for (int i = 0; i < 4; ++i) {
      v2h t0; t0[0] = (_Float16)p0[2 * i]; t0[1] = (_Float16)p0[2 * i + 1];
      v2h t1; t1[0] = (_Float16)p1[2 * i]; t1[1] = (_Float16)p1[2 * i + 1];
      int d0 = __builtin_bit_cast(int, t0);
      int d1 = __builtin_bit_cast(int, t1);
      int x0 = xor16i(d0);
      int x1 = xor16i(d1);
      up.i[i]     = (half == 0) ? d0 : x1;  // elements 2i,2i+1 : keys 16h+2i..
      up.i[4 + i] = (half == 0) ? x0 : d1;  // elements 8+2i    : keys 16h+8+2i..
    }
    v16h bP = up.v;

    // ---- O^T += V^T . P^T ----
#pragma unroll
    for (int c = 0; c < 4; ++c) {
      acc[c] *= alpha;
      v16h aV = ld_frag16(&ldsVt[cur][(c * 16 + col) * VSTRH + half * 16]);
      acc[c] = WMMA16(aV, bP, acc[c]);
    }
  }

  // ---- normalize and store: lane=query col, rows d = c*16 + 8*half + i ----
  float inv = 1.0f / lrun;
  float* orow = O + ((size_t)(b * Ll + q0 + col) * Hh + head) * Dd;
#pragma unroll
  for (int c = 0; c < 4; ++c) {
    int d0 = c * 16 + half * 8;
    f4 o0 = {acc[c][0] * inv, acc[c][1] * inv, acc[c][2] * inv, acc[c][3] * inv};
    f4 o1 = {acc[c][4] * inv, acc[c][5] * inv, acc[c][6] * inv, acc[c][7] * inv};
    *(f4*)(orow + d0)     = o0;
    *(f4*)(orow + d0 + 4) = o1;
  }
}

extern "C" void kernel_launch(void* const* d_in, const int* in_sizes, int n_in,
                              void* d_out, int out_size, void* d_ws, size_t ws_size,
                              hipStream_t stream) {
  (void)in_sizes; (void)n_in; (void)out_size; (void)d_ws; (void)ws_size;
  const float* Q = (const float*)d_in[0];
  const float* K = (const float*)d_in[1];
  const float* V = (const float*)d_in[2];
  const int*   M = (const int*)d_in[3];
  float*       O = (float*)d_out;

  dim3 grid(Bb * Hh * (Ll / 128));  // 512 blocks
  dim3 block(256);                  // 8 waves x 16 queries
  fa_fwd_wmma<<<grid, block, 0, stream>>>(Q, K, V, M, O);
}
